// GeoMix3_33440615367382
// MI455X (gfx1250) — matrix operand order repaired and checked by reading, not compiled, not run on hardware
//
#include <hip/hip_runtime.h>
#include <hip/hip_bf16.h>

#define HOPS 3
#define RES_W 0.5f
#define GRAPH_W 0.5f

typedef __bf16 bf16_t;
typedef bf16_t v16bf __attribute__((ext_vector_type(16)));
typedef bf16_t v8bf  __attribute__((ext_vector_type(8)));
typedef float  v8f   __attribute__((ext_vector_type(8)));

// ---------------- elementwise / setup kernels ----------------
__global__ void k_copy_f(const float* __restrict__ in, float* __restrict__ out, int n) {
  int i = blockIdx.x * blockDim.x + threadIdx.x;
  if (i < n) out[i] = in[i];
}

__global__ void k_fill_f(float* __restrict__ p, float v, int n) {
  int i = blockIdx.x * blockDim.x + threadIdx.x;
  if (i < n) p[i] = v;
}

__global__ void k_cvt_bf16(const float* __restrict__ a, bf16_t* __restrict__ b, int n) {
  int i = blockIdx.x * blockDim.x + threadIdx.x;
  if (i < n) b[i] = (bf16_t)a[i];
}

__global__ void k_deg_edges(const int* __restrict__ dst, float* __restrict__ deg, int E) {
  int e = blockIdx.x * blockDim.x + threadIdx.x;
  if (e < E) atomicAdd(&deg[dst[e]], 1.0f);
}

__global__ void k_dinv(float* __restrict__ deg, int n) {
  int i = blockIdx.x * blockDim.x + threadIdx.x;
  if (i < n) { float d = deg[i]; deg[i] = (d > 0.0f) ? rsqrtf(d) : 0.0f; }
}

__global__ void k_edgew(const int* __restrict__ src, const int* __restrict__ dst,
                        const float* __restrict__ dinv, float* __restrict__ w, int E) {
  int e = blockIdx.x * blockDim.x + threadIdx.x;
  if (e < E) w[e] = dinv[src[e]] * dinv[dst[e]];
}

__global__ void k_wself(const float* __restrict__ dinv, float* __restrict__ ws, int n) {
  int i = blockIdx.x * blockDim.x + threadIdx.x;
  if (i < n) { float d = dinv[i]; ws[i] = d * d; }
}

// transpose + convert nm 64x64 fp32 mats [m][r][c] -> bf16 dst[m*dstride + c*64 + r]
__global__ void k_wt(const float* __restrict__ W, bf16_t* __restrict__ Wt, int nm, int dstride) {
  int i = blockIdx.x * blockDim.x + threadIdx.x;
  int tot = nm << 12;
  if (i < tot) {
    int m = i >> 12; int rc = i & 4095; int r = rc >> 6, c = rc & 63;
    Wt[m * dstride + (c << 6) + r] = (bf16_t)W[i];
  }
}

// M is [64][97] fp32 (last col = column-sum of kk); Mt is [96][64] bf16 (transposed)
__global__ void k_cvt_M(const float* __restrict__ M, bf16_t* __restrict__ Mt) {
  int i = blockIdx.x * blockDim.x + threadIdx.x;
  if (i < 96 * 64) {
    int p = i >> 6, k = i & 63;
    Mt[(p << 6) + k] = (bf16_t)M[k * 97 + p];
  }
}

// ---- shared WMMA tile core: acc = A_tile(16x64) @ Bt_tile(16x64)^T, f32 accum ----
__device__ __forceinline__ v8f wmma_tile_64(const bf16_t* __restrict__ Ap,
                                            const bf16_t* __restrict__ Bp, int half) {
  v8f acc = {0.f, 0.f, 0.f, 0.f, 0.f, 0.f, 0.f, 0.f};
#pragma unroll
  for (int k0 = 0; k0 < 64; k0 += 32) {
    // A 16x32 bf16 layout: elems 0..7 -> K=8*half+j, 8..15 -> K=8*half+16+j
    v8bf alo = *(const v8bf*)(Ap + k0 + 8 * half);
    v8bf ahi = *(const v8bf*)(Ap + k0 + 16 + 8 * half);
    v16bf a = __builtin_shufflevector(alo, ahi, 0,1,2,3,4,5,6,7,8,9,10,11,12,13,14,15);
    // B 32x16 bf16 layout: elems 0..7 -> K=8*half+j, 8..15 -> K=8*half+16+j of column N=lane%16
    v8bf blo = *(const v8bf*)(Bp + k0 + 8 * half);
    v8bf bhi = *(const v8bf*)(Bp + k0 + 16 + 8 * half);
    v16bf b = __builtin_shufflevector(blo, bhi, 0,1,2,3,4,5,6,7,8,9,10,11,12,13,14,15);
    acc = __builtin_amdgcn_wmma_f32_16x16x32_bf16(false, a, false, b, (short)0, acc, false, false);
  }
  return acc;
}

// ---------------- fused Q/K WMMA GEMM ----------------
// Bt: [128][64] bf16 = [WQ^T ; WK^T]. cols 0-63 -> qs, 64-127 -> kk. One wave per 16x16 tile.
__global__ void k_gemmqk_wmma(const bf16_t* __restrict__ A, const bf16_t* __restrict__ Bt,
                              const float* __restrict__ bq, const float* __restrict__ bk,
                              float* __restrict__ qs, float* __restrict__ kk, int n) {
  const int tilesP = 8;                    // P = 128
  int lane = threadIdx.x & 31;
  int wave = (int)((blockIdx.x * blockDim.x + threadIdx.x) >> 5);
  int tilesM = (n + 15) >> 4;
  if (wave >= tilesM * tilesP) return;
  int rowT = wave / tilesP, colT = wave - rowT * tilesP;
  int m0 = rowT << 4;
  int half = lane >> 4;
  int lm = lane & 15;
  int arow = m0 + lm; if (arow >= n) arow = n - 1;
  int bcol = (colT << 4) + lm;             // 0..127
  v8f acc = wmma_tile_64(A + (size_t)arow * 64, Bt + (size_t)bcol * 64, half);

  // column -> output buffer / bias (wave-uniform split at colT==4)
  float* C = (bcol < 64) ? qs : kk;
  int col = bcol & 63;
  float bv = (bcol < 64) ? bq[col] : bk[col];
  int base = m0 + 8 * half;
  if (base + 7 < n) {                      // full tile: branchless stores
#pragma unroll
    for (int r = 0; r < 8; ++r) C[(size_t)(base + r) * 64 + col] = acc[r] + bv;
  } else {
#pragma unroll
    for (int r = 0; r < 8; ++r)
      if (base + r < n) C[(size_t)(base + r) * 64 + col] = acc[r] + bv;
  }
}

// ---------------- WMMA GEMM with attention epilogue ----------------
// acc = qs @ M (P=96);  out = (scale*acc + n*base) / denom[row]
__global__ void k_apply_wmma(const bf16_t* __restrict__ A, const bf16_t* __restrict__ Bt,
                             const float* __restrict__ ax, const float* __restrict__ ay,
                             const float* __restrict__ den, const float* __restrict__ sq,
                             const float* __restrict__ sk,
                             float* __restrict__ ox, float* __restrict__ oy, int n) {
  const int tilesP = 6;                    // P = 96
  int lane = threadIdx.x & 31;
  int wave = (int)((blockIdx.x * blockDim.x + threadIdx.x) >> 5);
  int tilesM = (n + 15) >> 4;
  if (wave >= tilesM * tilesP) return;
  int rowT = wave / tilesP, colT = wave - rowT * tilesP;
  int m0 = rowT << 4;
  int half = lane >> 4;
  int lm = lane & 15;
  int arow = m0 + lm; if (arow >= n) arow = n - 1;
  int bcol = (colT << 4) + lm;             // 0..95
  v8f acc = wmma_tile_64(A + (size_t)arow * 64, Bt + (size_t)bcol * 64, half);

  float sc = rsqrtf(*sq) * rsqrtf(*sk);
  float nn = (float)n;
  // column -> base/out buffer (wave-uniform split at colT==4)
  const float* Bsrc = (bcol < 64) ? ax : ay;
  float*       Odst = (bcol < 64) ? ox : oy;
  int stride = (bcol < 64) ? 64 : 32;
  int col    = (bcol < 64) ? bcol : (bcol - 64);
  int base = m0 + 8 * half;
  if (base + 7 < n) {
#pragma unroll
    for (int r = 0; r < 8; ++r) {
      int row = base + r;
      float v = (sc * acc[r] + nn * Bsrc[(size_t)row * stride + col]) / den[row];
      Odst[(size_t)row * stride + col] = v;
    }
  } else {
#pragma unroll
    for (int r = 0; r < 8; ++r) {
      int row = base + r;
      if (row < n) {
        float v = (sc * acc[r] + nn * Bsrc[(size_t)row * stride + col]) / den[row];
        Odst[(size_t)row * stride + col] = v;
      }
    }
  }
}

// ---------------- kk^T @ [ax | ay | 1] -> M[64][97] via LDS partials + atomics ----------------
#define KTX_ROWS 64
__global__ void k_ktx(const float* __restrict__ kk, const float* __restrict__ ax,
                      const float* __restrict__ ay, float* __restrict__ M, int n) {
  __shared__ float sK[KTX_ROWS * 64];
  __shared__ float sV[KTX_ROWS * 97];
  int r0 = blockIdx.x * KTX_ROWS;
  for (int t = threadIdx.x; t < KTX_ROWS * 64; t += blockDim.x) {
    int r = t >> 6, c = t & 63; int row = r0 + r;
    sK[t] = (row < n) ? kk[(size_t)row * 64 + c] : 0.0f;
  }
  for (int t = threadIdx.x; t < KTX_ROWS * 97; t += blockDim.x) {
    int r = t / 97, c = t - r * 97; int row = r0 + r;
    float v = 0.0f;
    if (row < n) {
      if (c < 64)      v = ax[(size_t)row * 64 + c];
      else if (c < 96) v = ay[(size_t)row * 32 + (c - 64)];
      else             v = 1.0f;
    }
    sV[t] = v;
  }
  __syncthreads();
  for (int p = threadIdx.x; p < 64 * 97; p += blockDim.x) {
    int k = p / 97, c = p - k * 97;
    float s = 0.0f;
    for (int r = 0; r < KTX_ROWS; ++r) s += sK[r * 64 + k] * sV[r * 97 + c];
    atomicAdd(&M[k * 97 + c], s);
  }
}

// ---------------- sum of squares of qs and kk -> scalars ----------------
__global__ void k_sumsq(const float* __restrict__ q, const float* __restrict__ k_,
                        float* __restrict__ sq, float* __restrict__ sk, int n) {
  __shared__ float s0[256];
  __shared__ float s1[256];
  int tid = threadIdx.x;
  float aq = 0.0f, ak = 0.0f;
  for (int i = blockIdx.x * blockDim.x + tid; i < n; i += gridDim.x * blockDim.x) {
    float a = q[i]; aq += a * a;
    float b = k_[i]; ak += b * b;
  }
  s0[tid] = aq; s1[tid] = ak;
  __syncthreads();
  for (int off = 128; off > 0; off >>= 1) {
    if (tid < off) { s0[tid] += s0[tid + off]; s1[tid] += s1[tid + off]; }
    __syncthreads();
  }
  if (tid == 0) { atomicAdd(sq, s0[0]); atomicAdd(sk, s1[0]); }
}

// ---------------- denom[i] = scale*(qs_i . ksum) + n ----------------
__global__ void k_denom(const float* __restrict__ qs, const float* __restrict__ M,
                        const float* __restrict__ sq, const float* __restrict__ sk,
                        float* __restrict__ den, int n) {
  int i = blockIdx.x * blockDim.x + threadIdx.x;
  if (i >= n) return;
  float sc = rsqrtf(*sq) * rsqrtf(*sk);
  float s = 0.0f;
#pragma unroll 8
  for (int k = 0; k < 64; ++k) s += qs[(size_t)i * 64 + k] * M[k * 97 + 96];
  den[i] = sc * s + (float)n;
}

// ---------------- SPMM: init with self-loop, then edge scatter ----------------
__global__ void k_scale_rows(const float* __restrict__ f, const float* __restrict__ s,
                             float* __restrict__ o, int total, int shift) {
  int i = blockIdx.x * blockDim.x + threadIdx.x;
  if (i < total) o[i] = s[i >> shift] * f[i];
}

__global__ void k_spmm_edges(const int* __restrict__ src, const int* __restrict__ dst,
                             const float* __restrict__ w,
                             const float* __restrict__ fx, const float* __restrict__ fy,
                             float* __restrict__ ox, float* __restrict__ oy, int E) {
  int wid = (int)((blockIdx.x * blockDim.x + threadIdx.x) >> 5);
  int lane = threadIdx.x & 31;
  if (wid >= E) return;
  int s = src[wid], d = dst[wid];
  float we = w[wid];
  // out[src] += w * feat[dst] : lane covers x-channels {2*lane, 2*lane+1} and y-channel lane
  float2 vx = *(const float2*)(fx + (size_t)d * 64 + 2 * lane);
  float  vy = fy[(size_t)d * 32 + lane];
  atomicAdd(&ox[(size_t)s * 64 + 2 * lane],     we * vx.x);
  atomicAdd(&ox[(size_t)s * 64 + 2 * lane + 1], we * vx.y);
  atomicAdd(&oy[(size_t)s * 32 + lane],         we * vy);
}

// ---------------- final mix: a = RES*a + (1-RES)*(GW*g + (1-GW)*o) ----------------
__global__ void k_mix(float* __restrict__ a, const float* __restrict__ g,
                      const float* __restrict__ o, int n) {
  int i = blockIdx.x * blockDim.x + threadIdx.x;
  if (i < n) {
    float nx = GRAPH_W * g[i] + (1.0f - GRAPH_W) * o[i];
    a[i] = RES_W * a[i] + (1.0f - RES_W) * nx;
  }
}

extern "C" void kernel_launch(void* const* d_in, const int* in_sizes, int n_in,
                              void* d_out, int out_size, void* d_ws, size_t ws_size,
                              hipStream_t stream) {
  const float* x  = (const float*)d_in[0];
  const float* y  = (const float*)d_in[1];
  const int*   ei = (const int*)d_in[2];
  const float* WQ = (const float*)d_in[3];
  const float* bQ = (const float*)d_in[4];
  const float* WK = (const float*)d_in[5];
  const float* bK = (const float*)d_in[6];

  const int n = in_sizes[0] / 64;
  const int E = in_sizes[2] / 2;
  const int* src = ei;
  const int* dst = ei + E;

  float* ax = (float*)d_out;
  float* ay = ax + (size_t)n * 64;

  // carve workspace
  char* p = (char*)d_ws;
  auto carve = [&](size_t bytes) -> char* {
    char* q = p;
    p += (bytes + 255) & ~(size_t)255;
    return q;
  };
  float*  deg  = (float*)carve((size_t)n * 4);
  float*  w    = (float*)carve((size_t)E * 4);
  float*  wsf  = (float*)carve((size_t)n * 4);
  bf16_t* Wt   = (bf16_t*)carve((size_t)HOPS * 8192 * 2);   // per hop: [WQ^T(64x64) ; WK^T(64x64)]
  bf16_t* axb  = (bf16_t*)carve((size_t)n * 64 * 2);
  float*  qs   = (float*)carve((size_t)n * 64 * 4);
  float*  kkv  = (float*)carve((size_t)n * 64 * 4);
  bf16_t* qsb  = (bf16_t*)carve((size_t)n * 64 * 2);
  float*  M    = (float*)carve(64 * 97 * 4);
  bf16_t* Mt   = (bf16_t*)carve(96 * 64 * 2);
  float*  den  = (float*)carve((size_t)n * 4);
  float*  scal = (float*)carve(2 * 4);
  float*  gx0  = (float*)carve((size_t)n * 64 * 4);
  float*  gx1  = (float*)carve((size_t)n * 64 * 4);
  float*  gy0  = (float*)carve((size_t)n * 32 * 4);
  float*  gy1  = (float*)carve((size_t)n * 32 * 4);
  float*  ox   = (float*)carve((size_t)n * 64 * 4);
  float*  oy   = (float*)carve((size_t)n * 32 * 4);

  const int B = 256;
  auto g1 = [&](long long total) -> int { return (int)((total + B - 1) / B); };

  // ---- static graph preprocessing ----
  k_copy_f<<<g1((long long)n * 64), B, 0, stream>>>(x, ax, n * 64);
  k_copy_f<<<g1((long long)n * 32), B, 0, stream>>>(y, ay, n * 32);
  k_fill_f<<<g1(n), B, 0, stream>>>(deg, 1.0f, n);           // self loop
  k_deg_edges<<<g1(E), B, 0, stream>>>(dst, deg, E);
  k_dinv<<<g1(n), B, 0, stream>>>(deg, n);                   // deg -> dinv in place
  k_edgew<<<g1(E), B, 0, stream>>>(src, dst, deg, w, E);
  k_wself<<<g1(n), B, 0, stream>>>(deg, wsf, n);
  // weights: WQ^T at hop*8192, WK^T at hop*8192 + 4096
  k_wt<<<g1(HOPS * 4096), B, 0, stream>>>(WQ, Wt, HOPS, 8192);
  k_wt<<<g1(HOPS * 4096), B, 0, stream>>>(WK, Wt + 4096, HOPS, 8192);

  const int tilesM = (n + 15) >> 4;
  for (int hop = 0; hop < HOPS; ++hop) {
    // fused qs/kk GEMM (WMMA, bf16 inputs, f32 accum, P=128)
    k_cvt_bf16<<<g1((long long)n * 64), B, 0, stream>>>(ax, axb, n * 64);
    long long waves128 = (long long)tilesM * 8;
    k_gemmqk_wmma<<<g1(waves128 * 32), B, 0, stream>>>(axb, Wt + hop * 8192,
                                                       bQ + hop * 64, bK + hop * 64,
                                                       qs, kkv, n);

    // global Frobenius norms
    k_fill_f<<<1, 32, 0, stream>>>(scal, 0.0f, 2);
    k_sumsq<<<512, B, 0, stream>>>(qs, kkv, scal, scal + 1, n * 64);

    // M = kk^T @ [ax | ay | 1]
    k_fill_f<<<g1(64 * 97), B, 0, stream>>>(M, 0.0f, 64 * 97);
    k_ktx<<<(n + KTX_ROWS - 1) / KTX_ROWS, B, 0, stream>>>(kkv, ax, ay, M, n);

    // denom and attention output (WMMA with fused epilogue)
    k_denom<<<g1(n), B, 0, stream>>>(qs, M, scal, scal + 1, den, n);
    k_cvt_bf16<<<g1((long long)n * 64), B, 0, stream>>>(qs, qsb, n * 64);
    k_cvt_M<<<g1(96 * 64), B, 0, stream>>>(M, Mt);
    long long waves96 = (long long)tilesM * 6;
    k_apply_wmma<<<g1(waves96 * 32), B, 0, stream>>>(qsb, Mt, ax, ay, den, scal, scal + 1, ox, oy, n);

    // 3 SPMM hops (self-loop folded into init)
    const float* inx = ax; const float* iny = ay;
    float* bx[2] = {gx0, gx1}; float* by[2] = {gy0, gy1};
    for (int t = 0; t < HOPS; ++t) {
      float* outx = bx[t & 1]; float* outy = by[t & 1];
      k_scale_rows<<<g1((long long)n * 64), B, 0, stream>>>(inx, wsf, outx, n * 64, 6);
      k_scale_rows<<<g1((long long)n * 32), B, 0, stream>>>(iny, wsf, outy, n * 32, 5);
      k_spmm_edges<<<g1((long long)E * 32), B, 0, stream>>>(src, dst, w, inx, iny, outx, outy, E);
      inx = outx; iny = outy;
    }

    // residual mix (in place in d_out)
    k_mix<<<g1((long long)n * 64), B, 0, stream>>>(ax, inx, ox, n * 64);
    k_mix<<<g1((long long)n * 32), B, 0, stream>>>(ay, iny, oy, n * 32);
  }
}
